// SegmentGenerator_35347580846650
// MI455X (gfx1250) — matrix operand order repaired and checked by reading, not compiled, not run on hardware
//
#include <hip/hip_runtime.h>
#include <hip/hip_bf16.h>

typedef __attribute__((ext_vector_type(16))) _Float16 v16h;
typedef __attribute__((ext_vector_type(8)))  float    v8f;

#define MODEL_DIM 128
#define NSAMP     32768
#define NBANDS    128
#define NFRAMES   128
#define RES       32
#define BATCH     8
#define WPACK_PER_LAYER (8 * 28 * 32 * 16)   // == 128*128*7 = 114688 halfs

// ---------------------------------------------------------------------------
// Repack conv weights [L,128,128,7] f32 -> f16 in WMMA A-fragment order:
//   wp[l][otile][kk][lane][e]  so each lane reads 16 contiguous halfs (32 B)
// per K-step.  K-order q = k*128 + i  (tap-major), 28 steps of 32.
// ---------------------------------------------------------------------------
__global__ void repack_weights(const float* __restrict__ w,
                               _Float16* __restrict__ wp) {
  const int l = blockIdx.y;
  int idx = blockIdx.x * blockDim.x + threadIdx.x;
  if (idx >= WPACK_PER_LAYER) return;
  const int e     = idx & 15;
  const int lane  = (idx >> 4) & 31;
  const int kk    = (idx >> 9) % 28;
  const int otile = idx / (28 * 512);
  const int m = lane & 15, hi = lane >> 4;
  const int k = kk >> 2, i0 = (kk & 3) * 32;
  // A 16x32 K order (matches B): lanes0-15 K=0..15, lanes16-31 K=16..31
  const int ka = e + (hi ? 16 : 0);
  const int o  = otile * 16 + m;
  wp[(size_t)l * WPACK_PER_LAYER + idx] =
      (_Float16)w[(((size_t)l * 128 + o) * 128 + (i0 + ka)) * 7 + k];
}

// ---------------------------------------------------------------------------
// Dense layer out[b,j] = bias[j] + sum_k z[b,k] W[j,k] as WMMA GEMM:
//   M = 16 (8 batch rows + 8 zero rows), N = 16 cols of J, K = 128 (4 steps).
// A fragments staged once in LDS (branch-free WMMA loop); B is contiguous
// W-row f32 -> cvt_pk_f16.
// ---------------------------------------------------------------------------
__global__ __launch_bounds__(256) void linear_wmma(
    const float* __restrict__ z,     // [8,128]
    const float* __restrict__ W,     // [J,128]
    const float* __restrict__ bias,  // [J]
    float* __restrict__ out, int J) {
  const int lane = threadIdx.x & 31;
  const int wave = threadIdx.x >> 5;
  const int tile = blockIdx.x * 8 + wave;   // column tile of 16
  const int m = lane & 15, hi = lane >> 4;
  const int j = tile * 16 + m;              // this lane's B column

  __shared__ _Float16 afrag[4][32][16];     // A fragments for all 4 K-steps
  for (int idx = threadIdx.x; idx < 4 * 32 * 16; idx += 256) {
    int e  = idx & 15;
    int ln = (idx >> 4) & 31;
    int kk = idx >> 9;
    int mm = ln & 15, hh = ln >> 4;
    int ka = e + (hh ? 16 : 0);
    float v = (mm < 8) ? z[mm * MODEL_DIM + kk * 32 + ka] : 0.0f;
    afrag[kk][ln][e] = (_Float16)v;
  }
  __syncthreads();

  v8f acc = {};
  for (int kk = 0; kk < 4; ++kk) {
    const int k0 = kk * 32;
    union { v16h v; uint4 q[2]; } ua;
    const uint4* ap = (const uint4*)(&afrag[kk][lane][0]);
    ua.q[0] = ap[0]; ua.q[1] = ap[1];
    v16h bm;
#pragma unroll
    for (int e = 0; e < 16; ++e) {
      int kb = e + (hi ? 16 : 0);           // contiguous along W row
      bm[e] = (_Float16)W[(size_t)j * MODEL_DIM + k0 + kb];
    }
    acc = __builtin_amdgcn_wmma_f32_16x16x32_f16(false, ua.v, false, bm,
                                                 (short)0, acc, false, false);
  }
  if (hi == 0) {                            // D rows 0..7 = batches
#pragma unroll
    for (int v = 0; v < 8; ++v)
      out[v * J + j] = acc[v] + bias[j];
  }
}

// ---------------------------------------------------------------------------
// One ConvUpsample layer: y = leaky0.2(conv1d_k7(upsample2(x), W) + b)
// WMMA f32_16x16x32_f16: M=16 out-ch (wave), N=16 positions (block), K=896.
// A from pre-packed f16 (2x b128/lane/step); B from position-major LDS
// (2x ds_b128/lane/step, 16B-aligned, row pad breaks bank conflicts).
// ---------------------------------------------------------------------------
__global__ __launch_bounds__(256) void conv_up_layer(
    const float* __restrict__ x,       // [8,128,Lin]
    float* __restrict__ y,             // [8,128,2*Lin]
    const _Float16* __restrict__ wp,   // packed [8,28,32,16]
    const float* __restrict__ bias,    // [128]
    int Lin) {
  const int Lout = Lin * 2;
  const int b    = blockIdx.y;
  const int t0   = blockIdx.x * 16;
  const int tid  = threadIdx.x;
  const int lane = tid & 31;
  const int wave = tid >> 5;                  // out-channel tile 0..7

  __shared__ _Float16 xu[24][136];            // [upsampled pos - (t0-3)][channel]

  // Stage x2-nearest-upsampled input (+3 halo each side, zero pad) as f16.
  for (int idx = tid; idx < 22 * 128; idx += 256) {
    int j = idx >> 7, i = idx & 127;
    int p = t0 - 3 + j;
    float v = 0.0f;
    if (p >= 0 && p < Lout) v = x[(b * 128 + i) * Lin + (p >> 1)];
    xu[j][i] = (_Float16)v;
  }
  __syncthreads();

  const int m  = lane & 15;                   // B column (time) / store column
  const int hi = lane >> 4;
  const _Float16* wbase = wp + (size_t)wave * 14336 + lane * 16;
  __builtin_prefetch(wbase, 0, 1);            // global_prefetch weight stream

  v8f acc = {};
  for (int kk = 0; kk < 28; ++kk) {
    const int k  = kk >> 2;                   // tap 0..6
    const int i0 = (kk & 3) * 32;             // in-channel block
    union { v16h v; uint4 q[2]; } ua, ub;
    const uint4* ap = (const uint4*)(wbase + kk * 512);
    ua.q[0] = ap[0]; ua.q[1] = ap[1];
    const uint4* bp = (const uint4*)(&xu[m + k][i0 + 16 * hi]);
    ub.q[0] = bp[0]; ub.q[1] = bp[1];
    acc = __builtin_amdgcn_wmma_f32_16x16x32_f16(false, ua.v, false, ub.v,
                                                 (short)0, acc, false, false);
  }

  // D layout: VGPR v -> row M = v + 8*hi ; column N = lane&15
  const int t = t0 + m;
  if (t < Lout) {
#pragma unroll
    for (int v = 0; v < 8; ++v) {
      int oc = wave * 16 + v + (hi ? 8 : 0);
      float val = acc[v] + bias[oc];
      val = val > 0.0f ? val : 0.2f * val;    // leaky relu
      y[(b * 128 + oc) * Lout + t] = val;
    }
  }
}

// ---------------------------------------------------------------------------
// Envelope head: 128->1 conv (K=7) then square -> orig_env [8,256]
// ---------------------------------------------------------------------------
__global__ void env_final(const float* __restrict__ x,   // [8,128,256]
                          const float* __restrict__ fw,  // [1,128,7]
                          const float* __restrict__ fb,
                          float* __restrict__ oenv) {    // [8,256]
  int idx = blockIdx.x * blockDim.x + threadIdx.x;
  if (idx >= BATCH * 256) return;
  int b = idx >> 8, t = idx & 255;
  float acc = fb[0];
  for (int k = 0; k < 7; ++k) {
    int p = t + k - 3;
    if (p < 0 || p >= 256) continue;
    for (int i = 0; i < 128; ++i) acc += fw[i * 7 + k] * x[(b * 128 + i) * 256 + p];
  }
  oenv[idx] = acc * acc;
}

// ---------------------------------------------------------------------------
// Transfer head fused with hard Gumbel and resonance table:
// one wave32 per (batch, band); lane r = resolution slot (wave32-native).
// ---------------------------------------------------------------------------
__global__ __launch_bounds__(256) void tr_final_gumbel_amp(
    const float* __restrict__ x,    // [8,128,128]
    const float* __restrict__ fw,   // [32,128,7]
    const float* __restrict__ fb,   // [32]
    const float* __restrict__ gu,   // [8,128,32]
    float* __restrict__ amp) {      // [8,128,128]
  int wave = threadIdx.x >> 5, lane = threadIdx.x & 31;
  int p = blockIdx.x * 8 + wave;
  int b = p >> 7, n = p & 127;
  int r = lane;

  float acc = fb[r];
  for (int k = 0; k < 7; ++k) {
    int pos = n + k - 3;
    if (pos < 0 || pos >= 128) continue;
    for (int i = 0; i < 128; ++i)
      acc += fw[(r * 128 + i) * 7 + k] * x[(b * 128 + i) * 128 + pos];
  }
  float u = gu[(b * 128 + n) * 32 + r];
  acc += -__logf(-__logf(u));

  float v = acc; int idx = r;                // argmax, first index wins ties
  for (int off = 16; off; off >>= 1) {
    float ov = __shfl_xor(v, off);
    int   oi = __shfl_xor(idx, off);
    if (ov > v || (ov == v && oi < idx)) { v = ov; idx = oi; }
  }
  float d = 0.99f * (float)idx / 31.0f;      // linspace(0,0.99,32)[argmax]
#pragma unroll
  for (int j = 0; j < 4; ++j) {
    int f = lane + 32 * j;
    amp[(b * 128 + n) * 128 + f] = powf(d, (float)(f + 1));
  }
}

// ---------------------------------------------------------------------------
// env_up[b,s] = lin_interp(orig_env, 256->32768) * noise[s]
// ---------------------------------------------------------------------------
__global__ void env_upsample(const float* __restrict__ oenv,
                             const float* __restrict__ noise,
                             float* __restrict__ eup) {
  int idx = blockIdx.x * blockDim.x + threadIdx.x;
  if (idx >= BATCH * NSAMP) return;
  int b = idx >> 15, s = idx & (NSAMP - 1);
  float src = ((float)s + 0.5f) * (256.0f / (float)NSAMP) - 0.5f;
  src = fminf(fmaxf(src, 0.0f), 255.0f);
  int lo = (int)floorf(src);
  int hi = min(lo + 1, 255);
  float w = src - (float)lo;
  float e = oenv[b * 256 + lo] * (1.0f - w) + oenv[b * 256 + hi] * w;
  eup[idx] = e * noise[s];
}

// ---------------------------------------------------------------------------
// Band-summed resonator bank (mean over bands pushed through the conv):
// tfs[b,s] = sum_n lin_interp(amp[b,n],128->32768)[s] * sin(2*pi*f_n*s/SR)
// ---------------------------------------------------------------------------
__global__ __launch_bounds__(256) void tf_sum_kernel(
    const float* __restrict__ amp, float* __restrict__ tfs) {
  __shared__ float freqs[NBANDS];
  if (threadIdx.x < NBANDS) {
    float m0 = 2595.0f * log10f(1.0f + 30.0f / 700.0f);
    float m1 = 2595.0f * log10f(1.0f + 11025.0f / 700.0f);
    float mm = m0 + (m1 - m0) * (float)threadIdx.x / 127.0f;
    freqs[threadIdx.x] = 700.0f * (__expf((mm / 2595.0f) * 2.302585093f) - 1.0f);
  }
  __syncthreads();
  int idx = blockIdx.x * blockDim.x + threadIdx.x;
  if (idx >= BATCH * NSAMP) return;
  int b = idx >> 15, s = idx & (NSAMP - 1);
  float src = ((float)s + 0.5f) * (128.0f / (float)NSAMP) - 0.5f;
  src = fminf(fmaxf(src, 0.0f), 127.0f);
  int lo = (int)floorf(src);
  int hic = min(lo + 1, 127);
  float w = src - (float)lo;
  float t = (float)s / 22050.0f;
  const float* ab = amp + b * NBANDS * NFRAMES;
  float sum = 0.0f;
  for (int n = 0; n < NBANDS; ++n) {
    float a = ab[n * NFRAMES + lo] * (1.0f - w) + ab[n * NFRAMES + hic] * w;
    sum += a * __sinf(6.28318530718f * freqs[n] * t);
  }
  tfs[idx] = sum;
}

// ---------------------------------------------------------------------------
// Truncated linear convolution + /128 (mean over 128 bands already folded):
//   out[b,s] = (1/128) * sum_{tau<=s} eup[b,tau] * tfs[b,s-tau]
// ---------------------------------------------------------------------------
__global__ __launch_bounds__(256) void conv_mean(
    const float* __restrict__ eup, const float* __restrict__ tfs,
    float* __restrict__ out) {
  __shared__ float e_s[256];
  __shared__ float t_s[512];
  const int b   = blockIdx.y;
  const int S0  = blockIdx.x * 256;
  const int tid = threadIdx.x;
  const float* E = eup + b * NSAMP;
  const float* T = tfs + b * NSAMP;
  float acc = 0.0f;
  for (int T0 = 0; T0 <= S0; T0 += 256) {
    __syncthreads();
    e_s[tid] = E[T0 + tid];
    const int base = S0 - T0 - 255;
    for (int j = tid; j < 512; j += 256) {
      int g = base + j;
      t_s[j] = (g >= 0 && g < NSAMP) ? T[g] : 0.0f;
    }
    __syncthreads();
#pragma unroll 8
    for (int tj = 0; tj < 256; ++tj)
      acc += e_s[tj] * t_s[tid + 255 - tj];
  }
  out[b * NSAMP + S0 + tid] = acc * (1.0f / 128.0f);
}

// ---------------------------------------------------------------------------
extern "C" void kernel_launch(void* const* d_in, const int* in_sizes, int n_in,
                              void* d_out, int out_size, void* d_ws, size_t ws_size,
                              hipStream_t stream) {
  const float* transfer = (const float*)d_in[0];
  const float* elw = (const float*)d_in[1];
  const float* elb = (const float*)d_in[2];
  const float* ecw = (const float*)d_in[3];   // [6,128,128,7]
  const float* ecb = (const float*)d_in[4];   // [6,128]
  const float* efw = (const float*)d_in[5];
  const float* efb = (const float*)d_in[6];
  const float* tlw = (const float*)d_in[7];
  const float* tlb = (const float*)d_in[8];
  const float* tcw = (const float*)d_in[9];   // [4,128,128,7]
  const float* tcb = (const float*)d_in[10];  // [4,128]
  const float* tfw = (const float*)d_in[11];
  const float* tfb = (const float*)d_in[12];
  const float* noise = (const float*)d_in[13];
  const float* gu    = (const float*)d_in[14];

  float* out_final = (float*)d_out;                // [8,1,32768]
  float* oenv      = out_final + BATCH * NSAMP;    // [8,1,256]

  float* bufA = (float*)d_ws;                      // 8*128*256
  float* bufB = bufA + BATCH * 128 * 256;
  float* amp  = bufB + BATCH * 128 * 256;          // 8*128*128
  float* eup  = amp  + BATCH * 128 * 128;          // 8*32768
  float* tfs  = eup  + BATCH * NSAMP;              // 8*32768
  _Float16* wpE = (_Float16*)(tfs + BATCH * NSAMP);          // 6 packed layers
  _Float16* wpT = wpE + (size_t)6 * WPACK_PER_LAYER;         // 4 packed layers

  // ---- pack all conv weights to f16 WMMA fragments ----
  repack_weights<<<dim3(WPACK_PER_LAYER / 256, 6), 256, 0, stream>>>(ecw, wpE);
  repack_weights<<<dim3(WPACK_PER_LAYER / 256, 4), 256, 0, stream>>>(tcw, wpT);

  // ---- envelope ConvUpsample (start=4, 6 layers -> length 256) ----
  linear_wmma<<<4, 256, 0, stream>>>(transfer, elw, elb, bufA, 512);
  {
    int Lin = 4; float* src = bufA; float* dst = bufB;
    for (int l = 0; l < 6; ++l) {
      int Lout = Lin * 2;
      dim3 grid((Lout + 15) / 16, BATCH);
      conv_up_layer<<<grid, 256, 0, stream>>>(src, dst,
          wpE + (size_t)l * WPACK_PER_LAYER, ecb + l * 128, Lin);
      float* t = src; src = dst; dst = t; Lin = Lout;
    }
    env_final<<<(BATCH * 256 + 255) / 256, 256, 0, stream>>>(src, efw, efb, oenv);
  }

  // ---- transfer-function ConvUpsample (start=8, 4 layers -> length 128) ----
  linear_wmma<<<8, 256, 0, stream>>>(transfer, tlw, tlb, bufA, 1024);
  {
    int Lin = 8; float* src = bufA; float* dst = bufB;
    for (int l = 0; l < 4; ++l) {
      int Lout = Lin * 2;
      dim3 grid((Lout + 15) / 16, BATCH);
      conv_up_layer<<<grid, 256, 0, stream>>>(src, dst,
          wpT + (size_t)l * WPACK_PER_LAYER, tcb + l * 128, Lin);
      float* t = src; src = dst; dst = t; Lin = Lout;
    }
    tr_final_gumbel_amp<<<128, 256, 0, stream>>>(src, tfw, tfb, gu, amp);
  }

  // ---- fused tail ----
  env_upsample<<<(BATCH * NSAMP + 255) / 256, 256, 0, stream>>>(oenv, noise, eup);
  tf_sum_kernel<<<(BATCH * NSAMP + 255) / 256, 256, 0, stream>>>(amp, tfs);
  conv_mean<<<dim3(NSAMP / 256, BATCH), 256, 0, stream>>>(eup, tfs, out_final);
}
